// VGNN_27049704030896
// MI455X (gfx1250) — matrix operand order repaired and checked by reading, not compiled, not run on hardware
//
#include <hip/hip_runtime.h>
#include <hip/hip_bf16.h>
#include <math.h>

// Problem constants (match reference)
constexpr int Bn   = 16;
constexpr int Ln   = 1024;
constexpr int Nn   = Ln + 1;      // 1025 nodes
constexpr int En   = 128;
constexpr int Hn   = 4;
constexpr int Dn   = 128;
constexpr int HEn  = Hn * En;     // 512
constexpr int NPAD = 1056;        // N padded to multiple of 32 (WMMA K-chunks)
constexpr int NT   = NPAD / 16;   // 66 row-tiles

#define ALPHAK 0.2f
#define EPSLN  1e-6f

typedef __attribute__((ext_vector_type(16))) __bf16 v16bf;
typedef __attribute__((ext_vector_type(8)))  __bf16 v8bf;
typedef __attribute__((ext_vector_type(8)))  float  v8f;

__device__ __forceinline__ float leakyf(float x) { return x >= 0.f ? x : ALPHAK * x; }
__device__ __forceinline__ float eluf(float x)   { return x > 0.f ? x : (__expf(x) - 1.f); }

__device__ __forceinline__ v16bf pack16(v8bf lo, v8bf hi) {
  v16bf r;
#pragma unroll
  for (int k = 0; k < 8; ++k) { r[k] = lo[k]; r[k + 8] = hi[k]; }
  return r;
}

__device__ __forceinline__ v8f wmma_bf16(v16bf a, v16bf b, v8f c) {
  // D(f32,16x16) = A(16x32 bf16) x B(32x16 bf16) + C
  return __builtin_amdgcn_wmma_f32_16x16x32_bf16(false, a, false, b, (short)0, c, false, false);
}

// ---------------------------------------------------------------------------
// K1: Wh[h,n,:] = emb[n,:] @ W_enc[h]  (emb row 0 zeroed, pad rows zero)
// Store fp32 (for f1/f2 dots) and row-major bf16 (WMMA B operand).
// ---------------------------------------------------------------------------
__global__ __launch_bounds__(256) void k_wh(const float* __restrict__ embed,
                                            const float* __restrict__ W_enc,
                                            float* __restrict__ Wh,
                                            __bf16* __restrict__ Wh_bf) {
  const int nt = blockIdx.x, h = blockIdx.y;
  const int n0 = nt * 16, tid = threadIdx.x;
  __shared__ float es[16 * En];
  for (int idx = tid; idx < 16 * En; idx += 256) {
    const int r = idx >> 7, k = idx & (En - 1);
    const int n = n0 + r;
    es[idx] = (n > 0 && n < Nn) ? embed[(size_t)n * En + k] : 0.f;
  }
  __syncthreads();
  const int c  = tid & (En - 1);
  const int rb = (tid >> 7) * 8;
  float acc[8] = {0.f, 0.f, 0.f, 0.f, 0.f, 0.f, 0.f, 0.f};
  for (int k = 0; k < En; ++k) {
    const float w = W_enc[((size_t)h * En + k) * En + c];
#pragma unroll
    for (int r = 0; r < 8; ++r) acc[r] += es[(rb + r) * En + k] * w;
  }
#pragma unroll
  for (int r = 0; r < 8; ++r) {
    const size_t o = ((size_t)h * NPAD + n0 + rb + r) * En + c;
    Wh[o]    = acc[r];
    Wh_bf[o] = (__bf16)acc[r];
  }
}

// K1b: f1/f2 attention score vectors (one thread per (h,n))
__global__ __launch_bounds__(256) void k_scores(const float* __restrict__ Wh,
                                                const float* __restrict__ a_src,
                                                const float* __restrict__ a_dst,
                                                float* __restrict__ f1,
                                                float* __restrict__ f2) {
  const int idx = blockIdx.x * 256 + threadIdx.x;
  if (idx >= Hn * NPAD) return;
  const int h = idx / NPAD;
  const float* w = Wh + (size_t)idx * En;
  float s1 = 0.f, s2 = 0.f;
  for (int f = 0; f < En; ++f) { s1 += w[f] * a_src[h * En + f]; s2 += w[f] * a_dst[h * En + f]; }
  f1[idx] = s1;
  f2[idx] = s2;
}

// K1c: bf16 cast (W_dec is already K-major [h][e][f], exactly the WMMA B layout)
__global__ __launch_bounds__(256) void k_cast_bf16(const float* __restrict__ src,
                                                   __bf16* __restrict__ dst, int n) {
  const int i = blockIdx.x * 256 + threadIdx.x;
  if (i < n) dst[i] = (__bf16)src[i];
}

// ---------------------------------------------------------------------------
// K2: fused encoder attention  enc_pre[b,n,h*E+f] = elu( softmax_row @ Wh )
// Rank-1 logits + exact precomputed rowmax -> single-pass flash-style loop.
// A-tile (16x32 probs, bf16) is generated in LDS each chunk; B from Wh_bf.
// 128 threads = 4 waves, each wave owns two 16x16 f-tiles.
// ---------------------------------------------------------------------------
__global__ __launch_bounds__(128) void k_enc_att(const int* __restrict__ data,
                                                 const float* __restrict__ f1,
                                                 const float* __restrict__ f2,
                                                 const __bf16* __restrict__ Wh_bf,
                                                 float* __restrict__ enc_pre) {
  const int it = blockIdx.x, h = blockIdx.y, b = blockIdx.z;
  const int i0 = it * 16;
  const int tid = threadIdx.x;
  const int lane = tid & 31, wave = tid >> 5;

  __shared__ __align__(16) float  msk[NPAD];   // additive mask: 0 or -1e30
  __shared__ __align__(16) float  f2s[NPAD];
  __shared__ __align__(16) __bf16 Psh[16 * 32];
  __shared__ float denom[16];
  __shared__ float rowmax[16];
  __shared__ float red[128];

  for (int j = tid; j < NPAD; j += 128) {
    msk[j] = (j < Ln && data[(size_t)b * Ln + j] != 0) ? 0.f : -1e30f;  // z_in pads with 0
    f2s[j] = f2[h * NPAD + j];
  }
  __syncthreads();

  // rowmax_i = leaky(f1_i + max_allowed f2)   (leaky monotone)
  float mx = -1e30f;
  for (int j = tid; j < NPAD; j += 128) mx = fmaxf(mx, f2s[j] + msk[j]);
  red[tid] = mx;
  __syncthreads();
  for (int s = 64; s > 0; s >>= 1) { if (tid < s) red[tid] = fmaxf(red[tid], red[tid + s]); __syncthreads(); }
  const float maxf2 = red[0];
  __syncthreads();
  if (tid < 16) rowmax[tid] = leakyf(f1[h * NPAD + i0 + tid] + maxf2);
  __syncthreads();

  // P-tile generation mapping: thread t -> row t>>3, cols (t&7)*4 .. +3
  const int   prow  = tid >> 3;
  const int   pcol0 = (tid & 7) * 4;
  const float f1row = f1[h * NPAD + i0 + prow];
  const float rmrow = rowmax[prow];

  // WMMA fragment lane mapping (CDNA5 ISA)
  const int half = lane >> 4;            // A: K-halves; C/D: M-halves
  const int c0   = half * 8;
  const int m    = lane & 15;
  const int kB   = m + 16 * half;        // unused; B operand lane picks column n = m
  const int f0a  = (2 * wave) * 16;
  const int f0b  = f0a + 16;

  v8f acc0 = {}; v8f acc1 = {};
  float dpart = 0.f;                     // partial softmax denominator (row prow)

  for (int jb = 0; jb < NPAD; jb += 32) {
#pragma unroll
    for (int q = 0; q < 4; ++q) {
      const int col = pcol0 + q;
      const int j   = jb + col;
      const float p = __expf(leakyf(f1row + f2s[j]) + msk[j] - rmrow);
      dpart += p;
      Psh[prow * 32 + col] = (__bf16)p;
    }
    __syncthreads();
    const __bf16* pr = &Psh[m * 32 + c0];
    const v16bf A = pack16(*(const v8bf*)pr, *(const v8bf*)(pr + 16));
    const __bf16* bp = Wh_bf + ((size_t)h * NPAD + jb + c0) * En + m;
    v16bf B0, B1;
#pragma unroll
    for (int s = 0; s < 8; ++s) {
      B0[s] = bp[(size_t)s * En + f0a]; B0[s + 8] = bp[(size_t)(s + 16) * En + f0a];
      B1[s] = bp[(size_t)s * En + f0b]; B1[s + 8] = bp[(size_t)(s + 16) * En + f0b];
    }
    acc0 = wmma_bf16(A, B0, acc0);
    acc1 = wmma_bf16(A, B1, acc1);
    __syncthreads();
  }

  red[tid] = dpart;
  __syncthreads();
  if (tid < 16) {
    float s = 0.f;
#pragma unroll
    for (int q = 0; q < 8; ++q) s += red[tid * 8 + q];
    denom[tid] = s;
  }
  __syncthreads();

#pragma unroll
  for (int r = 0; r < 8; ++r) {
    const int ir = r + 8 * half;         // C layout: M = r (+8 for upper lanes)
    const int i  = i0 + ir;
    const bool ok  = (msk[i] == 0.f) && (denom[ir] > 0.f);
    const float iv = ok ? 1.f / denom[ir] : 0.f;
    const float v0 = ok ? eluf(acc0[r] * iv) : 0.f;
    const float v1 = ok ? eluf(acc1[r] * iv) : 0.f;
    const size_t rowo = ((size_t)b * NPAD + i) * HEn + h * En;
    enc_pre[rowo + f0a + m] = v0;
    enc_pre[rowo + f0b + m] = v1;
  }
}

// ---------------------------------------------------------------------------
// K3: layernorm(512, ddof=1) + elu -> bf16 (WMMA A operand for decoder GEMM)
// ---------------------------------------------------------------------------
__global__ __launch_bounds__(128) void k_ln(const float* __restrict__ enc_pre,
                                            const float* __restrict__ g,
                                            const float* __restrict__ bta,
                                            __bf16* __restrict__ enc_bf) {
  const int bid = blockIdx.x;
  const int n   = bid % NPAD;
  const int tid = threadIdx.x;
  __shared__ float red[128];
  const size_t base = (size_t)bid * HEn;
  if (n >= Nn) {
#pragma unroll
    for (int q = 0; q < 4; ++q) enc_bf[base + tid + 128 * q] = (__bf16)0.f;
    return;
  }
  float x[4];
  float s = 0.f;
#pragma unroll
  for (int q = 0; q < 4; ++q) { x[q] = enc_pre[base + tid + 128 * q]; s += x[q]; }
  red[tid] = s; __syncthreads();
  for (int st = 64; st > 0; st >>= 1) { if (tid < st) red[tid] += red[tid + st]; __syncthreads(); }
  const float mu = red[0] / (float)HEn;
  __syncthreads();
  float v = 0.f;
#pragma unroll
  for (int q = 0; q < 4; ++q) { const float d = x[q] - mu; v += d * d; }
  red[tid] = v; __syncthreads();
  for (int st = 64; st > 0; st >>= 1) { if (tid < st) red[tid] += red[tid + st]; __syncthreads(); }
  const float sd  = sqrtf(red[0] / (float)(HEn - 1));
  const float inv = 1.f / (sd + EPSLN);
#pragma unroll
  for (int q = 0; q < 4; ++q) {
    const int c = tid + 128 * q;
    enc_bf[base + c] = (__bf16)eluf(g[c] * (x[q] - mu) * inv + bta[c]);
  }
}

// ---------------------------------------------------------------------------
// K4: Wh2[b,h,n,:] = enc[b,n,:] @ W_dec[h]   (K = 512, bf16 WMMA, fp32 out)
// ---------------------------------------------------------------------------
__global__ __launch_bounds__(128) void k_dec_gemm(const __bf16* __restrict__ enc_bf,
                                                  const __bf16* __restrict__ Wdec_bf,
                                                  float* __restrict__ Wh2) {
  const int nt = blockIdx.x, h = blockIdx.y, b = blockIdx.z;
  const int n0  = nt * 16;
  const int tid = threadIdx.x;
  const int lane = tid & 31, wave = tid >> 5;
  const int half = lane >> 4, c0 = half * 8, m = lane & 15;
  const int kB   = m + 16 * half;
  const int f0a  = (2 * wave) * 16, f0b = f0a + 16;
  const __bf16* ap = enc_bf + ((size_t)b * NPAD + n0 + m) * HEn;
  v8f acc0 = {}; v8f acc1 = {};
  for (int eb = 0; eb < HEn; eb += 32) {
    const v16bf A  = pack16(*(const v8bf*)(ap + eb + c0), *(const v8bf*)(ap + eb + c0 + 16));
    const __bf16* bp = Wdec_bf + ((size_t)h * HEn + eb + c0) * Dn + m;
    v16bf B0, B1;
#pragma unroll
    for (int s = 0; s < 8; ++s) {
      B0[s] = bp[(size_t)s * Dn + f0a]; B0[s + 8] = bp[(size_t)(s + 16) * Dn + f0a];
      B1[s] = bp[(size_t)s * Dn + f0b]; B1[s + 8] = bp[(size_t)(s + 16) * Dn + f0b];
    }
    acc0 = wmma_bf16(A, B0, acc0);
    acc1 = wmma_bf16(A, B1, acc1);
  }
#pragma unroll
  for (int r = 0; r < 8; ++r) {
    const int i = n0 + r + 8 * half;
    const size_t rowo = ((size_t)(b * Hn + h) * NPAD + i) * Dn;
    Wh2[rowo + f0a + m] = acc0[r];
    Wh2[rowo + f0b + m] = acc1[r];
  }
}

// ---------------------------------------------------------------------------
// K5: decoder attention, ONLY row N-1 needed -> weighted row-sum, mean over h
// ---------------------------------------------------------------------------
__global__ __launch_bounds__(256) void k_dec_att(const int* __restrict__ data,
                                                 const float* __restrict__ Wh2,
                                                 const float* __restrict__ a_src,
                                                 const float* __restrict__ a_dst,
                                                 float* __restrict__ dec_row) {
  const int b   = blockIdx.x;
  const int tid = threadIdx.x;
  __shared__ float msk[NPAD];
  __shared__ float g2s[NPAD];
  __shared__ float red[256];
  for (int j = tid; j < NPAD; j += 256) {
    float v = -1e30f;
    if (j < Ln)       v = (data[(size_t)b * Ln + j] != 0) ? 0.f : -1e30f;
    else if (j == Ln) v = 0.f;                       // z_out pads with 1
    msk[j] = v;
  }
  const int fl = tid & 127, jh = tid >> 7;
  float accT = 0.f;
  for (int h = 0; h < Hn; ++h) {
    __syncthreads();
    const float* Wb = Wh2 + ((size_t)(b * Hn + h) * NPAD) * Dn;
    for (int j = tid; j < NPAD; j += 256) {
      float s = 0.f;
      if (j < Nn) {
        const float* w = Wb + (size_t)j * Dn;
        for (int f = 0; f < Dn; ++f) s += w[f] * a_dst[h * Dn + f];
      }
      g2s[j] = s;
    }
    __syncthreads();
    red[tid] = (tid < Dn) ? Wb[(size_t)(Nn - 1) * Dn + tid] * a_src[h * Dn + tid] : 0.f;
    __syncthreads();
    for (int st = 128; st > 0; st >>= 1) { if (tid < st) red[tid] += red[tid + st]; __syncthreads(); }
    const float g1 = red[0];
    __syncthreads();
    float mxp = -1e30f;
    for (int j = tid; j < NPAD; j += 256) mxp = fmaxf(mxp, g2s[j] + msk[j]);
    red[tid] = mxp; __syncthreads();
    for (int st = 128; st > 0; st >>= 1) { if (tid < st) red[tid] = fmaxf(red[tid], red[tid + st]); __syncthreads(); }
    const float rm = leakyf(g1 + red[0]);
    __syncthreads();
    float dp = 0.f;
    for (int j = tid; j < NPAD; j += 256) {
      const float wv = __expf(leakyf(g1 + g2s[j]) + msk[j] - rm);
      g2s[j] = wv;
      dp += wv;
    }
    red[tid] = dp; __syncthreads();
    for (int st = 128; st > 0; st >>= 1) { if (tid < st) red[tid] += red[tid + st]; __syncthreads(); }
    const float dnm = red[0];
    __syncthreads();
    float part = 0.f;
    for (int j = jh; j < Nn; j += 2) part += g2s[j] * Wb[(size_t)j * Dn + fl];
    accT += part / dnm;
  }
  __syncthreads();
  red[tid] = accT; __syncthreads();
  if (tid < Dn) dec_row[(size_t)b * Dn + tid] = (red[tid] + red[tid + 128]) / (float)Hn;
}

// ---------------------------------------------------------------------------
// K6: LN(D,ddof=1)+relu -> @V_w+V_b -> @o1_w+o1_b,relu -> @o2_w+o2_b
// ---------------------------------------------------------------------------
__global__ __launch_bounds__(128) void k_head(const float* __restrict__ dec_row,
                                              const float* __restrict__ g,
                                              const float* __restrict__ bta,
                                              const float* __restrict__ V_w,
                                              const float* __restrict__ V_b,
                                              const float* __restrict__ o1_w,
                                              const float* __restrict__ o1_b,
                                              const float* __restrict__ o2_w,
                                              const float* __restrict__ o2_b,
                                              float* __restrict__ out) {
  const int b = blockIdx.x;
  const int tid = threadIdx.x;
  __shared__ float red[128];
  __shared__ float v[128];
  __shared__ float x2[128];
  const float x = dec_row[(size_t)b * Dn + tid];
  red[tid] = x; __syncthreads();
  for (int st = 64; st > 0; st >>= 1) { if (tid < st) red[tid] += red[tid + st]; __syncthreads(); }
  const float mu = red[0] / (float)Dn;
  __syncthreads();
  const float d = x - mu;
  red[tid] = d * d; __syncthreads();
  for (int st = 64; st > 0; st >>= 1) { if (tid < st) red[tid] += red[tid + st]; __syncthreads(); }
  const float sd = sqrtf(red[0] / (float)(Dn - 1));
  __syncthreads();
  v[tid] = fmaxf(g[tid] * d / (sd + EPSLN) + bta[tid], 0.f);
  __syncthreads();
  float s = V_b[tid];
  for (int k = 0; k < Dn; ++k) s += v[k] * V_w[k * Dn + tid];
  x2[tid] = s;
  __syncthreads();
  float hh = o1_b[tid];
  for (int k = 0; k < Dn; ++k) hh += x2[k] * o1_w[k * Dn + tid];
  hh = fmaxf(hh, 0.f);
  red[tid] = hh * o2_w[tid]; __syncthreads();
  for (int st = 64; st > 0; st >>= 1) { if (tid < st) red[tid] += red[tid + st]; __syncthreads(); }
  if (tid == 0) out[b] = red[0] + o2_b[0];
}

// ---------------------------------------------------------------------------
extern "C" void kernel_launch(void* const* d_in, const int* in_sizes, int n_in,
                              void* d_out, int out_size, void* d_ws, size_t ws_size,
                              hipStream_t stream) {
  (void)in_sizes; (void)n_in; (void)out_size; (void)ws_size;
  const int*   data  = (const int*)  d_in[0];
  const float* embed = (const float*)d_in[1];
  const float* W_enc = (const float*)d_in[2];
  const float* a_es  = (const float*)d_in[3];
  const float* a_ed  = (const float*)d_in[4];
  const float* g_enc = (const float*)d_in[5];
  const float* b_enc = (const float*)d_in[6];
  const float* W_dec = (const float*)d_in[7];
  const float* a_ds  = (const float*)d_in[8];
  const float* a_dd  = (const float*)d_in[9];
  const float* g_dec = (const float*)d_in[10];
  const float* b_dec = (const float*)d_in[11];
  const float* V_w   = (const float*)d_in[12];
  const float* V_b   = (const float*)d_in[13];
  const float* o1_w  = (const float*)d_in[14];
  const float* o1_b  = (const float*)d_in[15];
  const float* o2_w  = (const float*)d_in[16];
  const float* o2_b  = (const float*)d_in[17];
  float* out = (float*)d_out;

  char* ws = (char*)d_ws;
  size_t off = 0;
  auto carve = [&](size_t bytes) -> char* {
    char* p = ws + off;
    off = (off + bytes + 255) & ~(size_t)255;
    return p;
  };
  float*  Wh      = (float*) carve((size_t)Hn * NPAD * En * 4);
  __bf16* Wh_bf   = (__bf16*)carve((size_t)Hn * NPAD * En * 2);
  float*  f1      = (float*) carve((size_t)Hn * NPAD * 4);
  float*  f2      = (float*) carve((size_t)Hn * NPAD * 4);
  __bf16* Wdec_bf = (__bf16*)carve((size_t)Hn * HEn * Dn * 2);
  float*  encpre  = (float*) carve((size_t)Bn * NPAD * HEn * 4);
  __bf16* enc_bf  = (__bf16*)carve((size_t)Bn * NPAD * HEn * 2);
  float*  dec_r   = (float*) carve((size_t)Bn * Dn * 4);
  float*  Wh2     = encpre;  // enc_pre dead after k_ln -> alias as Wh2 (same size)

  k_wh<<<dim3(NT, Hn), 256, 0, stream>>>(embed, W_enc, Wh, Wh_bf);
  k_scores<<<(Hn * NPAD + 255) / 256, 256, 0, stream>>>(Wh, a_es, a_ed, f1, f2);
  k_cast_bf16<<<(Hn * HEn * Dn + 255) / 256, 256, 0, stream>>>(W_dec, Wdec_bf, Hn * HEn * Dn);
  k_enc_att<<<dim3(NT, Hn, Bn), 128, 0, stream>>>(data, f1, f2, Wh_bf, encpre);
  k_ln<<<Bn * NPAD, 128, 0, stream>>>(encpre, g_enc, b_enc, enc_bf);
  k_dec_gemm<<<dim3(NT, Hn, Bn), 128, 0, stream>>>(enc_bf, Wdec_bf, Wh2);
  k_dec_att<<<Bn, 256, 0, stream>>>(data, Wh2, a_ds, a_dd, dec_r);
  k_head<<<Bn, 128, 0, stream>>>(dec_r, g_dec, b_dec, V_w, V_b, o1_w, o1_b, o2_w, o2_b, out);
}
